// EdgeBlock_38345468018709
// MI455X (gfx1250) — compile-verified, matrix operand order checked by reading, not run on hardware
//
#include <hip/hip_runtime.h>

// ---------------------------------------------------------------------------
// EdgeBlock MLP (GNN edge update) for MI455X / gfx1250.
//   x  = [edge_attr | node[row] | node[col]]            (E x 384, f32)
//   h1 = relu(x @ W_in + b_in)                          (E x 128)
//   h2 = relu(h1 @ W_h + b_h)                           (E x 128)
//   y  = LayerNorm(h2 @ W_out + b_out) * gamma + beta   (E x 128, f32)
//
// Roofline: 98 GFLOP, ~640 MB HBM (27 us @ 23.3 TB/s). Weights live in LDS
// (loaded once per block from a pre-swizzled bf16 copy in d_ws); M=32 rows
// per wave so each B fragment feeds two WMMAs. v3: 512-thread blocks
// (16 waves, 512 edges, 296 KB LDS of the 320 KB WGP budget) for 4 waves/SIMD
// occupancy, plus depth-2 software pipelining of B-fragment ds_loads so WMMAs
// overlap LDS latency. bf16 WMMA 16x16x32, f32 accumulate.
// ---------------------------------------------------------------------------

#define N_EDGES 600000
#define N_NODES 50000

typedef __attribute__((ext_vector_type(16))) __bf16 v16bf;
typedef __attribute__((ext_vector_type(8)))  __bf16 v8bf;
typedef __attribute__((ext_vector_type(8)))  float  v8f;

#define W1_ELEMS (12*8*32*16)   /* 49152: W_in  fragments [kt][nt][lane][16] */
#define W2_ELEMS (4*8*32*16)    /* 16384: W_h   fragments */
#define W3_ELEMS (4*8*32*16)    /* 16384: W_out fragments */
#define W_TOTAL  (W1_ELEMS + W2_ELEMS + W3_ELEMS)   /* 81920 elems = 160 KB */

// Pre-swizzle f32 weights -> bf16 fragments matching the CDNA5 16-bit
// B-matrix (32x16 per WMMA) lane layout (CDNA5 ISA 7.12.2):
//   lane L: n = nt*16 + (L%16), h = L/16
//   element e (0..15): K = kt*32 + (e<8 ? h*8+e : 16 + h*8 + (e-8))
__global__ __launch_bounds__(256) void swizzle_weights_kernel(
    const float* __restrict__ W_in, const float* __restrict__ W_h,
    const float* __restrict__ W_out, __bf16* __restrict__ ws)
{
  int t = blockIdx.x * blockDim.x + threadIdx.x;
  if (t >= W_TOTAL) return;
  const float* W; int loc;
  if (t < W1_ELEMS)                { W = W_in;  loc = t; }
  else if (t < W1_ELEMS + W2_ELEMS){ W = W_h;   loc = t - W1_ELEMS; }
  else                             { W = W_out; loc = t - (W1_ELEMS + W2_ELEMS); }
  int e    = loc & 15;
  int lane = (loc >> 4) & 31;
  int nt   = (loc >> 9) & 7;
  int kt   = loc >> 12;
  int h    = lane >> 4;
  int K    = kt*32 + ((e < 8) ? (h*8 + e) : (16 + h*8 + (e - 8)));
  int n    = nt*16 + (lane & 15);
  ws[t] = (__bf16)W[K*128 + n];
}

__global__ __launch_bounds__(512) void edge_mlp_wmma_kernel(
    const float* __restrict__ edge_attr,
    const float* __restrict__ node_attr,
    const long long* __restrict__ edge_index,
    const float* __restrict__ b_in,
    const float* __restrict__ b_h,
    const float* __restrict__ b_out,
    const float* __restrict__ ln_gamma,
    const float* __restrict__ ln_beta,
    const __bf16* __restrict__ wfrag,
    float* __restrict__ out)
{
  // All weight fragments, block-resident (160 KB).
  __shared__ __align__(16) __bf16 lds_w[W_TOTAL];
  // Per-wave 32x128 bf16 activation scratch for the D->A reshuffle between
  // layers (stride 136 keeps 16B alignment & staggers banks). 136 KB.
  __shared__ __align__(16) __bf16 scratch[16][32][136];

  const int tid  = threadIdx.x;
  const int w    = tid >> 5;        // wave id (16 waves)
  const int lane = tid & 31;
  const int l16  = lane & 15;
  const int h    = lane >> 4;       // half-wave select
  const int e_base = blockIdx.x * 512 + w * 32;   // 32 edges per wave

  // ---- cooperative copy of pre-swizzled weights into LDS (32B chunks) ----
  {
    const v16bf* src = (const v16bf*)wfrag;
    v16bf*       dst = (v16bf*)lds_w;
    #pragma unroll
    for (int i = 0; i < W_TOTAL/16/512; ++i)       // 10 iterations
      dst[i*512 + tid] = src[i*512 + tid];
  }
  __syncthreads();

  // A-fragment rows owned by this lane (row m = lane%16) for both M-tiles.
  // Clamp loads at the tail so EXEC stays all-ones for WMMA; stores are
  // predicated instead.
  int e0 = e_base + l16;
  int e1 = e_base + 16 + l16;
  int c0i = (e0 < N_EDGES) ? e0 : (N_EDGES - 1);
  int c1i = (e1 < N_EDGES) ? e1 : (N_EDGES - 1);
  long long s0 = edge_index[c0i];
  long long r0 = edge_index[(long long)N_EDGES + c0i];
  long long s1 = edge_index[c1i];
  long long r1 = edge_index[(long long)N_EDGES + c1i];

  const float* ap0[3] = { edge_attr + (long long)c0i * 128,
                          node_attr + s0 * 128, node_attr + r0 * 128 };
  const float* ap1[3] = { edge_attr + (long long)c1i * 128,
                          node_attr + s1 * 128, node_attr + r1 * 128 };

  v8f acc0[8] = {};   // tile0: rows e_base..+15
  v8f acc1[8] = {};   // tile1: rows e_base+16..+31

  const __bf16* w1 = lds_w;
  const __bf16* w2 = lds_w + W1_ELEMS;
  const __bf16* w3 = lds_w + W1_ELEMS + W2_ELEMS;

  // B-fragment fetch: fragment f (f = kt*8+nt) is 512 elems; lane reads a
  // contiguous 32B chunk -> 2x ds_load_b128, conflict-free.
  #define BFRAG(wb, f) (*(const v16bf*)((wb) + (f)*512 + lane*16))

  // ------------------------- Layer 1: x[32,384] @ W_in -------------------
  {
    v16bf bA = BFRAG(w1, 0);
    v16bf bB = BFRAG(w1, 1);
    #pragma unroll
    for (int kt = 0; kt < 12; ++kt) {
      const int reg = kt >> 2, off = (kt & 3) * 32;
      const float* p0 = ap0[reg] + off;
      const float* p1 = ap1[reg] + off;
      // A layout: lane holds K = h*8+e (e<8) and K = 16+h*8+(e-8).
      v8f q0 = *(const v8f*)(p0 + h*8);
      v8f q1 = *(const v8f*)(p0 + 16 + h*8);
      v8f q2 = *(const v8f*)(p1 + h*8);
      v8f q3 = *(const v8f*)(p1 + 16 + h*8);
      v16bf a0 = __builtin_shufflevector(__builtin_convertvector(q0, v8bf),
                                         __builtin_convertvector(q1, v8bf),
          0,1,2,3,4,5,6,7,8,9,10,11,12,13,14,15);
      v16bf a1 = __builtin_shufflevector(__builtin_convertvector(q2, v8bf),
                                         __builtin_convertvector(q3, v8bf),
          0,1,2,3,4,5,6,7,8,9,10,11,12,13,14,15);
      #pragma unroll
      for (int nt = 0; nt < 8; ++nt) {
        const int f = kt*8 + nt;
        v16bf bcur = bA;
        bA = bB;
        if (f + 2 < 96) bB = BFRAG(w1, f + 2);   // prefetch 2 ahead
        acc0[nt] = __builtin_amdgcn_wmma_f32_16x16x32_bf16(
            false, a0, false, bcur, (short)0, acc0[nt], false, false);
        acc1[nt] = __builtin_amdgcn_wmma_f32_16x16x32_bf16(
            false, a1, false, bcur, (short)0, acc1[nt], false, false);
      }
    }
  }

  // bias + relu -> bf16 scratch (C/D layout: row m = r + 8*h, col nt*16+l16)
  #pragma unroll
  for (int nt = 0; nt < 8; ++nt) {
    float bb = b_in[nt*16 + l16];
    #pragma unroll
    for (int r = 0; r < 8; ++r) {
      float v0 = acc0[nt][r] + bb; v0 = v0 > 0.f ? v0 : 0.f;
      float v1 = acc1[nt][r] + bb; v1 = v1 > 0.f ? v1 : 0.f;
      scratch[w][r + 8*h][nt*16 + l16]      = (__bf16)v0;
      scratch[w][16 + r + 8*h][nt*16 + l16] = (__bf16)v1;
    }
    v8f z = {}; acc0[nt] = z; acc1[nt] = z;
  }

  // ------------------------- Layer 2: h1[32,128] @ W_h -------------------
  {
    v16bf bA = BFRAG(w2, 0);
    v16bf bB = BFRAG(w2, 1);
    #pragma unroll
    for (int kt = 0; kt < 4; ++kt) {
      const __bf16* p0 = &scratch[w][l16][kt*32];
      const __bf16* p1 = &scratch[w][16 + l16][kt*32];
      v16bf a0 = __builtin_shufflevector(*(const v8bf*)(p0 + h*8),
                                         *(const v8bf*)(p0 + 16 + h*8),
          0,1,2,3,4,5,6,7,8,9,10,11,12,13,14,15);
      v16bf a1 = __builtin_shufflevector(*(const v8bf*)(p1 + h*8),
                                         *(const v8bf*)(p1 + 16 + h*8),
          0,1,2,3,4,5,6,7,8,9,10,11,12,13,14,15);
      #pragma unroll
      for (int nt = 0; nt < 8; ++nt) {
        const int f = kt*8 + nt;
        v16bf bcur = bA;
        bA = bB;
        if (f + 2 < 32) bB = BFRAG(w2, f + 2);
        acc0[nt] = __builtin_amdgcn_wmma_f32_16x16x32_bf16(
            false, a0, false, bcur, (short)0, acc0[nt], false, false);
        acc1[nt] = __builtin_amdgcn_wmma_f32_16x16x32_bf16(
            false, a1, false, bcur, (short)0, acc1[nt], false, false);
      }
    }
  }

  #pragma unroll
  for (int nt = 0; nt < 8; ++nt) {
    float bb = b_h[nt*16 + l16];
    #pragma unroll
    for (int r = 0; r < 8; ++r) {
      float v0 = acc0[nt][r] + bb; v0 = v0 > 0.f ? v0 : 0.f;
      float v1 = acc1[nt][r] + bb; v1 = v1 > 0.f ? v1 : 0.f;
      scratch[w][r + 8*h][nt*16 + l16]      = (__bf16)v0;
      scratch[w][16 + r + 8*h][nt*16 + l16] = (__bf16)v1;
    }
    v8f z = {}; acc0[nt] = z; acc1[nt] = z;
  }

  // ------------------------- Layer 3: h2[32,128] @ W_out -----------------
  {
    v16bf bA = BFRAG(w3, 0);
    v16bf bB = BFRAG(w3, 1);
    #pragma unroll
    for (int kt = 0; kt < 4; ++kt) {
      const __bf16* p0 = &scratch[w][l16][kt*32];
      const __bf16* p1 = &scratch[w][16 + l16][kt*32];
      v16bf a0 = __builtin_shufflevector(*(const v8bf*)(p0 + h*8),
                                         *(const v8bf*)(p0 + 16 + h*8),
          0,1,2,3,4,5,6,7,8,9,10,11,12,13,14,15);
      v16bf a1 = __builtin_shufflevector(*(const v8bf*)(p1 + h*8),
                                         *(const v8bf*)(p1 + 16 + h*8),
          0,1,2,3,4,5,6,7,8,9,10,11,12,13,14,15);
      #pragma unroll
      for (int nt = 0; nt < 8; ++nt) {
        const int f = kt*8 + nt;
        v16bf bcur = bA;
        bA = bB;
        if (f + 2 < 32) bB = BFRAG(w3, f + 2);
        acc0[nt] = __builtin_amdgcn_wmma_f32_16x16x32_bf16(
            false, a0, false, bcur, (short)0, acc0[nt], false, false);
        acc1[nt] = __builtin_amdgcn_wmma_f32_16x16x32_bf16(
            false, a1, false, bcur, (short)0, acc1[nt], false, false);
      }
    }
  }

  // --------------------- bias + LayerNorm + store ------------------------
  float sum0[8] = {}, sq0[8] = {}, sum1[8] = {}, sq1[8] = {};
  #pragma unroll
  for (int nt = 0; nt < 8; ++nt) {
    float bb = b_out[nt*16 + l16];
    #pragma unroll
    for (int r = 0; r < 8; ++r) {
      float v0 = acc0[nt][r] + bb; acc0[nt][r] = v0;
      sum0[r] += v0; sq0[r] += v0 * v0;
      float v1 = acc1[nt][r] + bb; acc1[nt][r] = v1;
      sum1[r] += v1; sq1[r] += v1 * v1;
    }
  }
  // reduce across the 16 lanes of each half-wave (row m = r + 8*h)
  #pragma unroll
  for (int r = 0; r < 8; ++r) {
    #pragma unroll
    for (int off = 1; off < 16; off <<= 1) {
      sum0[r] += __shfl_xor(sum0[r], off, 32);
      sq0[r]  += __shfl_xor(sq0[r],  off, 32);
      sum1[r] += __shfl_xor(sum1[r], off, 32);
      sq1[r]  += __shfl_xor(sq1[r],  off, 32);
    }
  }
  float mu0[8], rs0[8], mu1[8], rs1[8];
  #pragma unroll
  for (int r = 0; r < 8; ++r) {
    mu0[r] = sum0[r] * (1.0f / 128.0f);
    float va0 = sq0[r] * (1.0f / 128.0f) - mu0[r] * mu0[r];
    rs0[r] = rsqrtf(va0 + 1e-5f);
    mu1[r] = sum1[r] * (1.0f / 128.0f);
    float va1 = sq1[r] * (1.0f / 128.0f) - mu1[r] * mu1[r];
    rs1[r] = rsqrtf(va1 + 1e-5f);
  }
  #pragma unroll
  for (int nt = 0; nt < 8; ++nt) {
    float g  = ln_gamma[nt*16 + l16];
    float bt = ln_beta[nt*16 + l16];
    #pragma unroll
    for (int r = 0; r < 8; ++r) {
      int ea = e_base + r + 8*h;
      int eb = e_base + 16 + r + 8*h;
      if (ea < N_EDGES)
        out[(long long)ea * 128 + nt*16 + l16] =
            (acc0[nt][r] - mu0[r]) * rs0[r] * g + bt;
      if (eb < N_EDGES)
        out[(long long)eb * 128 + nt*16 + l16] =
            (acc1[nt][r] - mu1[r]) * rs1[r] * g + bt;
    }
  }
  #undef BFRAG
}

extern "C" void kernel_launch(void* const* d_in, const int* in_sizes, int n_in,
                              void* d_out, int out_size, void* d_ws, size_t ws_size,
                              hipStream_t stream) {
  const float*     edge_attr  = (const float*)d_in[0];
  const float*     node_attr  = (const float*)d_in[1];
  const long long* edge_index = (const long long*)d_in[2];   // int64
  const float*     W_in  = (const float*)d_in[3];
  const float*     b_in  = (const float*)d_in[4];
  const float*     W_h   = (const float*)d_in[5];
  const float*     b_h   = (const float*)d_in[6];
  const float*     W_out = (const float*)d_in[7];
  const float*     b_out = (const float*)d_in[8];
  const float*     g     = (const float*)d_in[9];
  const float*     bt    = (const float*)d_in[10];

  __bf16* wfrag = (__bf16*)d_ws;   // 160 KB of scratch used

  swizzle_weights_kernel<<<(W_TOTAL + 255) / 256, 256, 0, stream>>>(
      W_in, W_h, W_out, wfrag);

  dim3 grid((N_EDGES + 511) / 512);   // 512 edges per block, 32 per wave
  edge_mlp_wmma_kernel<<<grid, 512, 0, stream>>>(
      edge_attr, node_attr, edge_index, b_in, b_h, b_out, g, bt,
      wfrag, (float*)d_out);
}
